// VQVAE_12799002542788
// MI455X (gfx1250) — matrix operand order repaired and checked by reading, not compiled, not run on hardware
//
#include <hip/hip_runtime.h>

// ---------------------------------------------------------------------------
// Multi-scale residual VQ (VAR VectorQuantizer2 hot path) for gfx1250.
// Distance search: v_wmma_f32_16x16x32_f16 with A = -2*z (f16), B = codebook
// (f16, LDS-staged via global_load_async_to_lds_b128 double buffering),
// C initialized to ||e||^2  =>  D = squared distance (up to +||z||^2 per row,
// constant w.r.t. the argmin).
// ---------------------------------------------------------------------------

typedef __attribute__((ext_vector_type(16))) _Float16 v16h;
typedef __attribute__((ext_vector_type(8)))  float    v8f;

#define NSCALES 10
#define B_      64
#define C_      32
#define HW_     256     // 16*16
#define V_      4096
#define NTILE_V 256     // 4096 / 16 code tiles
#define CHUNK_T 32      // code tiles per LDS chunk (32 KB)
#define NCHUNK  8       // 256 / 32

// ---------------- compile-time area / bicubic matrices --------------------
struct Tables { float A[1088]; float U[1088]; };  // sum(pn)*16 = 68*16

constexpr double cubicw(double x) {
  const double a = -0.75;
  double ax = x < 0 ? -x : x;
  if (ax <= 1.0) return (a + 2) * ax * ax * ax - (a + 3) * ax * ax + 1.0;
  if (ax < 2.0)  return a * ax * ax * ax - 5 * a * ax * ax + 8 * a * ax - 4 * a;
  return 0.0;
}

constexpr Tables make_tables() {
  Tables t{};
  const int pns[NSCALES] = {1, 2, 3, 4, 5, 6, 8, 10, 13, 16};
  int off = 0;
  for (int s = 0; s < NSCALES; ++s) {
    const int pn = pns[s];
    for (int i = 0; i < pn; ++i) {                 // area: [pn][16]
      const int lo = (i * 16) / pn;
      const int hi = ((i + 1) * 16 + pn - 1) / pn;
      for (int h = lo; h < hi; ++h) t.A[off + i * 16 + h] = 1.0f / (float)(hi - lo);
    }
    for (int i = 0; i < 16; ++i) {                 // bicubic: [16][pn]
      const double src = (i + 0.5) * (double)pn / 16.0 - 0.5;
      int i0 = (int)src; if (src < (double)i0) --i0;
      const double tt = src - (double)i0;
      for (int k = -1; k < 3; ++k) {
        int j = i0 + k; if (j < 0) j = 0; if (j > pn - 1) j = pn - 1;
        t.U[off + i * pn + j] += (float)cubicw((double)k - tt);
      }
    }
    off += pn * 16;
  }
  return t;
}

__constant__ Tables g_tab = make_tables();

// ---------------- prep: f_rest = f, f_hat = 0 ------------------------------
__global__ __launch_bounds__(256)
void vq_prep_state(const float* __restrict__ f, float* __restrict__ f_rest,
                   float* __restrict__ f_hat) {
  const int i = blockIdx.x * 256 + threadIdx.x;
  f_rest[i] = f[i];
  f_hat[i]  = 0.0f;
}

// ---------------- prep: pack codebook f16 (WMMA B layout) + ||e||^2 --------
// B operand 32x16 f16, contiguous k order chosen here (A matches): lanes 0-15 hold N=lane, K=0..15; lanes 16-31
// hold N=lane-16, K=16..31. Per lane: 16 contiguous halves per code tile.
__global__ __launch_bounds__(256)
void vq_prep_code(const float* __restrict__ emb, _Float16* __restrict__ ebuf,
                  float* __restrict__ e2) {
  const int g    = blockIdx.x * 256 + threadIdx.x;  // 0 .. 256*32-1
  const int t    = g >> 5;
  const int lane = g & 31;
  const int m    = lane & 15;
  const int half = lane >> 4;
  const int n    = t * 16 + m;
  const float* er = emb + n * C_;
  _Float16* dst = ebuf + (size_t)(t * 32 + lane) * 16;
#pragma unroll
  for (int j = 0; j < 16; ++j) dst[j] = (_Float16)er[half * 16 + j];
  if (half == 0) {
    float s = 0.f;
#pragma unroll
    for (int k = 0; k < C_; ++k) s += er[k] * er[k];
    e2[n] = s;
  }
}

// ---------------- separable area downsample: f_rest -> z (NHWC) ------------
__global__ __launch_bounds__(256)
void vq_down(const float* __restrict__ f_rest, int pn, int aoff,
             float* __restrict__ z) {
  __shared__ float s0[C_ * HW_];
  __shared__ float s1[C_ * HW_];
  const int b = blockIdx.x, tid = threadIdx.x;
  const float* A = g_tab.A + aoff;     // [pn][16]

  for (int i = tid; i < C_ * HW_; i += 256) s0[i] = f_rest[b * C_ * HW_ + i];
  __syncthreads();

  for (int i = tid; i < C_ * pn * 16; i += 256) {        // vertical pool
    const int w = i & 15, p = (i >> 4) % pn, c = (i >> 4) / pn;
    float s = 0.f;
#pragma unroll
    for (int h = 0; h < 16; ++h) s += A[p * 16 + h] * s0[c * HW_ + h * 16 + w];
    s1[(c * pn + p) * 16 + w] = s;
  }
  __syncthreads();

  for (int i = tid; i < C_ * pn * pn; i += 256) {        // horizontal pool
    const int c = i & 31, q = (i >> 5) % pn, p = (i >> 5) / pn;
    float s = 0.f;
#pragma unroll
    for (int w = 0; w < 16; ++w) s += A[q * 16 + w] * s1[(c * pn + p) * 16 + w];
    z[((size_t)(b * pn + p) * pn + q) * C_ + c] = s;
  }
}

// ---------------- WMMA codebook argmin + gather ----------------------------
// One wave handles a tile of 16 z-vectors against all 4096 codes. The packed
// f16 codebook is streamed through LDS with double-buffered async copies
// (ASYNCcnt-tracked), 32 KB per chunk; all 4 waves of the block share it.
__global__ __launch_bounds__(128)
void vq_quant(const float* __restrict__ z,        // [Nvec][32]
              const _Float16* __restrict__ ebuf,  // [256][32][16] halves
              const float* __restrict__ e2,       // [4096]
              const float* __restrict__ emb,      // [4096][32]
              float* __restrict__ h) {            // [Nvec][32]
  __shared__ _Float16 sB[2][CHUNK_T * 32 * 16];   // 2 x 32 KB double buffer
  __shared__ int sidx[4][16];

  const int tid  = threadIdx.x;
  const int lane = tid & 31;
  const int wid  = tid >> 5;
  const int tile = blockIdx.x * 4 + wid;
  const int m    = lane & 15;
  const int half = lane >> 4;
  const int vec  = tile * 16 + m;

  // ---- pack A = -2 * z (folds the -2 of the distance expansion) ----
  const float4* zv = (const float4*)(z + (size_t)vec * C_);
  const float4 p0 = zv[half * 4 + 0];
  const float4 p1 = zv[half * 4 + 1];
  const float4 p2 = zv[half * 4 + 2];
  const float4 p3 = zv[half * 4 + 3];
  v16h a;
  a[0]  = (_Float16)(-2.0f * p0.x); a[1]  = (_Float16)(-2.0f * p0.y);
  a[2]  = (_Float16)(-2.0f * p0.z); a[3]  = (_Float16)(-2.0f * p0.w);
  a[4]  = (_Float16)(-2.0f * p1.x); a[5]  = (_Float16)(-2.0f * p1.y);
  a[6]  = (_Float16)(-2.0f * p1.z); a[7]  = (_Float16)(-2.0f * p1.w);
  a[8]  = (_Float16)(-2.0f * p2.x); a[9]  = (_Float16)(-2.0f * p2.y);
  a[10] = (_Float16)(-2.0f * p2.z); a[11] = (_Float16)(-2.0f * p2.w);
  a[12] = (_Float16)(-2.0f * p3.x); a[13] = (_Float16)(-2.0f * p3.y);
  a[14] = (_Float16)(-2.0f * p3.z); a[15] = (_Float16)(-2.0f * p3.w);

  float bestV[8]; int bestN[8];
#pragma unroll
  for (int r = 0; r < 8; ++r) { bestV[r] = 3.4e38f; bestN[r] = 0; }

  const char* gbase = (const char*)ebuf;
  const unsigned l0 = (unsigned)(size_t)(&sB[0][0]);
  const unsigned l1 = (unsigned)(size_t)(&sB[1][0]);

  // ---- async prologue: chunk 0 -> buf 0 (16 x b128 per thread) ----
  {
    const char* g = gbase + tid * 16;
    const unsigned l = l0 + tid * 16;
#pragma unroll
    for (int i = 0; i < 16; ++i) {
      asm volatile("global_load_async_to_lds_b128 %0, %1, off"
                   :: "v"(l + i * 2048), "v"(g + (size_t)i * 2048) : "memory");
    }
  }

  for (int ch = 0; ch < NCHUNK; ++ch) {
    const int buf = ch & 1;
    if (ch + 1 < NCHUNK) {   // issue next chunk while previous finishes
      const char* g = gbase + (size_t)(ch + 1) * 32768 + tid * 16;
      const unsigned l = (buf ? l0 : l1) + tid * 16;
#pragma unroll
      for (int i = 0; i < 16; ++i) {
        asm volatile("global_load_async_to_lds_b128 %0, %1, off"
                     :: "v"(l + i * 2048), "v"(g + (size_t)i * 2048) : "memory");
      }
      asm volatile("s_wait_asynccnt 0x10" ::: "memory");  // chunk ch landed
    } else {
      asm volatile("s_wait_asynccnt 0x0" ::: "memory");
    }
    __syncthreads();

    const _Float16* Bp = &sB[buf][lane * 16];
#pragma unroll 2
    for (int tt = 0; tt < CHUNK_T; ++tt) {
      const int t = ch * CHUNK_T + tt;
      const v16h bmat = *(const v16h*)(Bp + tt * 512);   // (tt*32+lane)*16
      const int n = t * 16 + m;                          // this lane's column
      const float e2v = e2[n];
      v8f c = {e2v, e2v, e2v, e2v, e2v, e2v, e2v, e2v};
      c = __builtin_amdgcn_wmma_f32_16x16x32_f16(false, a, false, bmat,
                                                 (short)0, c, false, false);
#pragma unroll
      for (int r = 0; r < 8; ++r) {
        const int take = (int)(c[r] < bestV[r]);
        bestV[r] = take ? c[r] : bestV[r];
        bestN[r] = take ? n    : bestN[r];
      }
    }
    __syncthreads();   // chunk buffer free for reuse
  }

  // ---- branch-free lexicographic argmin over the 16 lanes of each half ----
#pragma unroll
  for (int r = 0; r < 8; ++r) {
    float v = bestV[r]; int n = bestN[r];
#pragma unroll
    for (int s = 1; s < 16; s <<= 1) {
      const float ov = __shfl_xor(v, s, 32);
      const int   on = __shfl_xor(n, s, 32);
      const int take = ((int)(ov < v)) | (((int)(ov == v)) & ((int)(on < n)));
      v = take ? ov : v;
      n = take ? on : n;
    }
    bestV[r] = v; bestN[r] = n;
  }
  if (m == 0) {
#pragma unroll
    for (int r = 0; r < 8; ++r) sidx[wid][half * 8 + r] = bestN[r];
  }
  __syncthreads();

  // ---- cooperative gather: lane == channel ----
  float* hrow = h + (size_t)tile * 16 * C_;
  for (int mm = 0; mm < 16; ++mm) {
    const int n = sidx[wid][mm];
    hrow[mm * C_ + lane] = emb[(size_t)n * C_ + lane];
  }
}

// ---------------- bicubic upsample + Phi conv + residual update ------------
__global__ __launch_bounds__(256)
void vq_refine(const float* __restrict__ hsm,   // [B][pn][pn][32]
               int pn, int uoff,
               const float* __restrict__ w,     // [32][32][3][3]
               const float* __restrict__ bias,  // [32]
               float* __restrict__ f_hat, float* __restrict__ f_rest) {
  __shared__ float s0[C_ * HW_];
  __shared__ float s1[C_ * HW_];
  const int b = blockIdx.x, tid = threadIdx.x;
  const float* U = g_tab.U + uoff;               // [16][pn]

  for (int i = tid; i < pn * pn * C_; i += 256) {      // stage [c][p*pn+q]
    const int c = i & 31, vec = i >> 5;
    s0[c * pn * pn + vec] = hsm[(size_t)(b * pn * pn + vec) * C_ + c];
  }
  __syncthreads();

  for (int i = tid; i < C_ * pn * 16; i += 256) {      // horizontal bicubic
    const int x = i & 15, p = (i >> 4) % pn, c = (i >> 4) / pn;
    float s = 0.f;
    for (int q = 0; q < pn; ++q) s += U[x * pn + q] * s0[c * pn * pn + p * pn + q];
    s1[(c * pn + p) * 16 + x] = s;
  }
  __syncthreads();

  for (int i = tid; i < C_ * HW_; i += 256) {          // vertical bicubic
    const int x = i & 15, y = (i >> 4) & 15, c = i >> 8;
    float s = 0.f;
    for (int p = 0; p < pn; ++p) s += U[y * pn + p] * s1[(c * pn + p) * 16 + x];
    s0[c * HW_ + y * 16 + x] = s;
  }
  __syncthreads();

  // 3x3 conv (zero pad) + Phi blend + residual update
  const int co  = tid >> 3;        // 32 output channels
  const int seg = tid & 7;         // 8 pixel segments of 32 (2 rows each)
  float acc[32];
#pragma unroll
  for (int p = 0; p < 32; ++p) acc[p] = 0.f;

  for (int ci = 0; ci < C_; ++ci) {
    float w9[9];
#pragma unroll
    for (int j = 0; j < 9; ++j) w9[j] = w[(size_t)(co * C_ + ci) * 9 + j];
    const float* hc = &s0[ci * HW_];
#pragma unroll 4
    for (int p = 0; p < 32; ++p) {
      const int px = seg * 32 + p;
      const int y = px >> 4, x = px & 15;
      float s = 0.f;
#pragma unroll
      for (int dy = -1; dy <= 1; ++dy) {
        const int yy = y + dy; if (yy < 0 || yy > 15) continue;
#pragma unroll
        for (int dx = -1; dx <= 1; ++dx) {
          const int xx = x + dx; if (xx < 0 || xx > 15) continue;
          s += w9[(dy + 1) * 3 + (dx + 1)] * hc[yy * 16 + xx];
        }
      }
      acc[p] += s;
    }
  }

  const float bv = bias[co];
#pragma unroll
  for (int p = 0; p < 32; ++p) {
    const int px  = seg * 32 + p;
    const float hh = s0[co * HW_ + px];
    const float out = 0.5f * hh + 0.5f * (acc[p] + bv);   // RESI = 0.5
    const size_t gi = (size_t)(b * C_ + co) * HW_ + px;
    f_hat[gi]  += out;
    f_rest[gi] -= out;
  }
}

// ---------------- host: Phi bank selection (ties -> first, like argmin) ----
static int phi_select_host(int si) {
  int bk = 0; double best = 1e30;
  for (int k = 0; k < 4; ++k) {
    const double tick = 1.0 / 12.0 + (double)k * (10.0 / 12.0) / 3.0;
    double d = tick - (double)si / 9.0; if (d < 0) d = -d;
    if (d < best - 1e-12) { best = d; bk = k; }
  }
  return bk;
}

extern "C" void kernel_launch(void* const* d_in, const int* in_sizes, int n_in,
                              void* d_out, int out_size, void* d_ws, size_t ws_size,
                              hipStream_t stream) {
  (void)in_sizes; (void)n_in; (void)out_size; (void)ws_size;
  const float* f     = (const float*)d_in[0];   // [64,32,16,16]
  const float* emb   = (const float*)d_in[1];   // [4096,32]
  const float* phi_w = (const float*)d_in[2];   // [4,32,32,3,3]
  const float* phi_b = (const float*)d_in[3];   // [4,32]
  float* f_hat = (float*)d_out;                 // [64,32,16,16]

  const size_t NF = (size_t)B_ * C_ * HW_;      // 524288
  float*    f_rest = (float*)d_ws;
  float*    zbuf   = f_rest + NF;
  float*    hbuf   = zbuf + NF;
  float*    e2     = hbuf + NF;
  _Float16* ebuf   = (_Float16*)(e2 + V_);

  vq_prep_state<<<(int)(NF / 256), 256, 0, stream>>>(f, f_rest, f_hat);
  vq_prep_code<<<32, 256, 0, stream>>>(emb, ebuf, e2);

  static const int pns[NSCALES] = {1, 2, 3, 4, 5, 6, 8, 10, 13, 16};
  int off = 0;
  for (int si = 0; si < NSCALES; ++si) {
    const int pn = pns[si];
    const int k  = phi_select_host(si);
    vq_down<<<B_, 256, 0, stream>>>(f_rest, pn, off, zbuf);
    vq_quant<<<pn * pn, 128, 0, stream>>>(zbuf, ebuf, e2, emb, hbuf);
    vq_refine<<<B_, 256, 0, stream>>>(hbuf, pn, off,
                                      phi_w + (size_t)k * C_ * C_ * 9,
                                      phi_b + (size_t)k * C_,
                                      f_hat, f_rest);
    off += pn * 16;
  }
}
